// PretrainDPS_51453708206796
// MI455X (gfx1250) — matrix operand order repaired and checked by reading, not compiled, not run on hardware
//
#include <hip/hip_runtime.h>
#include <hip/hip_bf16.h>

#define N_PIX 100352LL   // 8*112*112
#define HW    12544LL    // 112*112

typedef __attribute__((ext_vector_type(16))) __bf16 v16bf;
typedef __attribute__((ext_vector_type(8)))  float  v8f;
typedef __attribute__((ext_vector_type(8)))  unsigned int v8u;

union FragU { v8u u; v16bf b; };

// ---- CDNA5 async global->LDS path (guarded; falls back to ds_store staging)
#ifdef __has_builtin
#if __has_builtin(__builtin_amdgcn_global_load_async_to_lds_b128) && \
    __has_builtin(__builtin_amdgcn_s_wait_asynccnt)
#define USE_ASYNC_LDS 1
#endif
#endif
#ifndef USE_ASYNC_LDS
#define USE_ASYNC_LDS 0
#endif

#if USE_ASYNC_LDS
typedef int v4i32 __attribute__((vector_size(16)));   // matches builtin pointee type
__device__ inline void async_cp16(const void* g, void* l) {
  // AS1 address == generic global address; AS3 offset == low 32 bits of
  // the generic LDS address (flat LDS aperture keeps offset in [31:0]).
  __builtin_amdgcn_global_load_async_to_lds_b128(
      (__attribute__((address_space(1))) v4i32*)(unsigned long long)g,
      (__attribute__((address_space(3))) v4i32*)(unsigned int)(unsigned long long)l,
      0, 0);
}
#endif

__device__ inline unsigned short f2bf(float f) {
  unsigned int u = __float_as_uint(f);
  u += 0x7FFFu + ((u >> 16) & 1u);       // round-to-nearest-even
  return (unsigned short)(u >> 16);
}

__device__ inline float reduceN(float v, float* red, int t, int n) {
  red[t] = v; __syncthreads();
  for (int off = n >> 1; off > 0; off >>= 1) {
    if (t < off) red[t] += red[t + off];
    __syncthreads();
  }
  float r = red[0]; __syncthreads();
  return r;
}

// ---------------------------------------------------------------- utilities
__global__ void zero_kernel(float* p, int n) {
  int i = blockIdx.x * 256 + threadIdx.x;
  if (i < n) p[i] = 0.0f;
}

__global__ void cvt_bf16_kernel(unsigned short* dst, const float* src, int n) {
  int i = blockIdx.x * 256 + threadIdx.x;
  if (i < n) dst[i] = f2bf(src[i]);
}

__global__ __launch_bounds__(128) void protos_norm_kernel(
    const float* __restrict__ protos, float* __restrict__ protosn) {
  __shared__ float red[128];
  const int j = blockIdx.x, t = threadIdx.x;
  float v[4]; float ss = 0.f;
#pragma unroll
  for (int i = 0; i < 4; i++) { v[i] = protos[j * 512 + t + i * 128]; ss += v[i] * v[i]; }
  ss = reduceN(ss, red, t, 128);
  float rn = 1.f / fmaxf(sqrtf(ss), 1e-12f);
#pragma unroll
  for (int i = 0; i < 4; i++) protosn[j * 512 + t + i * 128] = v[i] * rn;
}

// -------------------------------------------- depthwise 3x3 over virtual concat
__global__ __launch_bounds__(256) void dw_kernel(
    const float* __restrict__ x1, const float* __restrict__ x2,
    const float* __restrict__ x3, const float* __restrict__ x4,
    const float* __restrict__ dww, unsigned short* __restrict__ out) {
  const int p = blockIdx.x * 32 + threadIdx.x;          // pixel
  const int c = blockIdx.y * 8 + threadIdx.y;           // channel 0..959
  const int n = p / (int)HW;
  const int rem = p % (int)HW;
  const int h = rem / 112, w = rem % 112;

  const float* src; int ch, sh, Cs;
  if (c < 64)       { src = x1; ch = c;       sh = 0; Cs = 64;  }
  else if (c < 192) { src = x2; ch = c - 64;  sh = 1; Cs = 128; }
  else if (c < 448) { src = x3; ch = c - 192; sh = 2; Cs = 256; }
  else              { src = x4; ch = c - 448; sh = 3; Cs = 512; }
  const int Hs = 112 >> sh;
  const float* plane = src + (long long)(n * Cs + ch) * Hs * Hs;

  float acc = 0.f;
#pragma unroll
  for (int dy = 0; dy < 3; dy++) {
#pragma unroll
    for (int dx = 0; dx < 3; dx++) {
      const int hh = h + dy - 1, ww = w + dx - 1;
      float v = 0.f;
      if (hh >= 0 && hh < 112 && ww >= 0 && ww < 112)
        v = plane[(hh >> sh) * Hs + (ww >> sh)];
      acc += dww[c * 9 + dy * 3 + dx] * v;
    }
  }
  out[(long long)p * 960 + c] = f2bf(acc);
}

// ---------------------------------------------------------- WMMA bf16 GEMM
// C[N x 512] = A[N x K] * B^T (B stored [n][k] row-major, n = output col)
// MODE 0: out = acc*w0[col] + w1[col]  -> bf16
// MODE 1: out = relu(acc + w1[col])    -> bf16
// MODE 2: out = acc + w1[col]          -> f32
template <int MODE>
__global__ __launch_bounds__(256) void gemm_bf16_kernel(
    const unsigned short* __restrict__ A, const unsigned short* __restrict__ B,
    int K, const float* __restrict__ w0, const float* __restrict__ w1,
    unsigned short* __restrict__ outB, float* __restrict__ outF) {
  const int t = threadIdx.x;
  const int lane = t & 31, wave = t >> 5;
  const int wm = wave & 3, wn = wave >> 2;      // 4x2 wave grid
  const int half = lane >> 4, l16 = lane & 15;
  const long long rowStart = (long long)blockIdx.x * 128;
  const int colStart = blockIdx.y * 64;

  const v8f vzero = {};
  v8f acc[2][2];
  acc[0][0] = vzero; acc[0][1] = vzero; acc[1][0] = vzero; acc[1][1] = vzero;

#if USE_ASYNC_LDS
  // double-buffered LDS, filled by async global->LDS copies (ASYNCcnt)
  __shared__ __attribute__((aligned(64))) unsigned short Alds[2][128 * 32];
  __shared__ __attribute__((aligned(64))) unsigned short Blds[2][64 * 32];

  const int r0 = t >> 2, sub = t & 3;   // A chunks: rows r0 and r0+64; B: col r0
  const int nt = K / 32;

#define STAGE_TILE(bufi, kk)                                                        \
  do {                                                                              \
    async_cp16(&A[(rowStart + r0) * K + (kk) + sub * 8],                            \
               &Alds[bufi][r0 * 32 + sub * 8]);                                     \
    async_cp16(&A[(rowStart + r0 + 64) * K + (kk) + sub * 8],                       \
               &Alds[bufi][(r0 + 64) * 32 + sub * 8]);                              \
    async_cp16(&B[(long long)(colStart + r0) * K + (kk) + sub * 8],                 \
               &Blds[bufi][r0 * 32 + sub * 8]);                                     \
  } while (0)

  STAGE_TILE(0, 0);
  for (int it = 0; it < nt; ++it) {
    if (it + 1 < nt) {
      STAGE_TILE((it + 1) & 1, (it + 1) * 32);   // stream next tile
      __builtin_amdgcn_s_wait_asynccnt(3);       // tile `it` landed (in-order)
    } else {
      __builtin_amdgcn_s_wait_asynccnt(0);
    }
    __syncthreads();
    const unsigned short* Ab = Alds[it & 1];
    const unsigned short* Bb = Blds[it & 1];

    FragU fa[2], fb[2];
#pragma unroll
    for (int ti = 0; ti < 2; ++ti) {
      const int m = wm * 32 + ti * 16 + l16;
      const uint4 lo = *(const uint4*)&Ab[m * 32 + half * 8];
      const uint4 hi = *(const uint4*)&Ab[m * 32 + 16 + half * 8];
      fa[ti].u[0] = lo.x; fa[ti].u[1] = lo.y; fa[ti].u[2] = lo.z; fa[ti].u[3] = lo.w;
      fa[ti].u[4] = hi.x; fa[ti].u[5] = hi.y; fa[ti].u[6] = hi.z; fa[ti].u[7] = hi.w;
    }
#pragma unroll
    for (int tj = 0; tj < 2; ++tj) {
      const int nn = wn * 32 + tj * 16 + l16;
      const uint4 lo = *(const uint4*)&Bb[nn * 32 + half * 8];
      const uint4 hi = *(const uint4*)&Bb[nn * 32 + 16 + half * 8];
      fb[tj].u[0] = lo.x; fb[tj].u[1] = lo.y; fb[tj].u[2] = lo.z; fb[tj].u[3] = lo.w;
      fb[tj].u[4] = hi.x; fb[tj].u[5] = hi.y; fb[tj].u[6] = hi.z; fb[tj].u[7] = hi.w;
    }
#pragma unroll
    for (int ti = 0; ti < 2; ++ti)
#pragma unroll
      for (int tj = 0; tj < 2; ++tj)
        acc[ti][tj] = __builtin_amdgcn_wmma_f32_16x16x32_bf16(
            false, fa[ti].b, false, fb[tj].b, (short)0, acc[ti][tj], false, false);
    __syncthreads();
  }
#undef STAGE_TILE
#else
  __shared__ __attribute__((aligned(64))) unsigned short Alds[128 * 32];
  __shared__ __attribute__((aligned(64))) unsigned short Blds[64 * 32];

  for (int k0 = 0; k0 < K; k0 += 32) {
#pragma unroll
    for (int i = 0; i < 2; ++i) {
      const int cch = t + i * 256;
      const int r = cch >> 2, sub = cch & 3;
      const uint4 v = *(const uint4*)&A[(rowStart + r) * K + k0 + sub * 8];
      *(uint4*)&Alds[r * 32 + sub * 8] = v;
    }
    {
      const int nn = t >> 2, sub = t & 3;
      const uint4 v = *(const uint4*)&B[(long long)(colStart + nn) * K + k0 + sub * 8];
      *(uint4*)&Blds[nn * 32 + sub * 8] = v;
    }
    if (k0 + 32 < K)   // warm caches for next A tile (global_prefetch_b8)
      __builtin_prefetch((const void*)&A[(rowStart + (t >> 2)) * K + k0 + 32], 0, 0);
    __syncthreads();

    FragU fa[2], fb[2];
#pragma unroll
    for (int ti = 0; ti < 2; ++ti) {
      const int m = wm * 32 + ti * 16 + l16;
      const uint4 lo = *(const uint4*)&Alds[m * 32 + half * 8];
      const uint4 hi = *(const uint4*)&Alds[m * 32 + 16 + half * 8];
      fa[ti].u[0] = lo.x; fa[ti].u[1] = lo.y; fa[ti].u[2] = lo.z; fa[ti].u[3] = lo.w;
      fa[ti].u[4] = hi.x; fa[ti].u[5] = hi.y; fa[ti].u[6] = hi.z; fa[ti].u[7] = hi.w;
    }
#pragma unroll
    for (int tj = 0; tj < 2; ++tj) {
      const int nn = wn * 32 + tj * 16 + l16;
      const uint4 lo = *(const uint4*)&Blds[nn * 32 + half * 8];
      const uint4 hi = *(const uint4*)&Blds[nn * 32 + 16 + half * 8];
      fb[tj].u[0] = lo.x; fb[tj].u[1] = lo.y; fb[tj].u[2] = lo.z; fb[tj].u[3] = lo.w;
      fb[tj].u[4] = hi.x; fb[tj].u[5] = hi.y; fb[tj].u[6] = hi.z; fb[tj].u[7] = hi.w;
    }
#pragma unroll
    for (int ti = 0; ti < 2; ++ti)
#pragma unroll
      for (int tj = 0; tj < 2; ++tj)
        acc[ti][tj] = __builtin_amdgcn_wmma_f32_16x16x32_bf16(
            false, fa[ti].b, false, fb[tj].b, (short)0, acc[ti][tj], false, false);
    __syncthreads();
  }
#endif

  // epilogue: D lane layout: row = 8*half + r, col = l16
#pragma unroll
  for (int ti = 0; ti < 2; ++ti) {
#pragma unroll
    for (int tj = 0; tj < 2; ++tj) {
      const int col = colStart + wn * 32 + tj * 16 + l16;
#pragma unroll
      for (int r = 0; r < 8; ++r) {
        const long long row = rowStart + wm * 32 + ti * 16 + half * 8 + r;
        float v = acc[ti][tj][r];
        if (MODE == 0) v = v * w0[col] + w1[col];
        else           v = v + w1[col];
        if (MODE == 1) v = v > 0.f ? v : 0.f;
        if (MODE == 2) outF[row * 512 + col] = v;
        else           outB[row * 512 + col] = f2bf(v);
      }
    }
  }
}

// ------------------------------- fused LN + L2 + masks + seg-LN + pred/correct
__global__ __launch_bounds__(128) void ln_masks_seg_kernel(
    float* __restrict__ c, const float* __restrict__ ln1g, const float* __restrict__ ln1b,
    const float* __restrict__ protosn, float* __restrict__ masks,
    float* __restrict__ outseg, const float* __restrict__ ln2g, const float* __restrict__ ln2b,
    const int* __restrict__ gt, float* __restrict__ correct) {
  __shared__ float buf[512];
  __shared__ float red[128];
  __shared__ float mk[20];
  const long long p = blockIdx.x;
  const int t = threadIdx.x;

  float x[4]; float s = 0.f, ss = 0.f;
#pragma unroll
  for (int i = 0; i < 4; i++) { x[i] = c[p * 512 + t + i * 128]; s += x[i]; ss += x[i] * x[i]; }
  s  = reduceN(s,  red, t, 128);
  ss = reduceN(ss, red, t, 128);
  const float mean = s * (1.f / 512.f);
  const float var  = ss * (1.f / 512.f) - mean * mean;
  const float rstd = rsqrtf(var + 1e-5f);
  float l2 = 0.f;
#pragma unroll
  for (int i = 0; i < 4; i++) {
    const int d = t + i * 128;
    const float y = (x[i] - mean) * rstd * ln1g[d] + ln1b[d];
    x[i] = y; l2 += y * y;
  }
  l2 = reduceN(l2, red, t, 128);
  const float rn = 1.f / fmaxf(sqrtf(l2), 1e-12f);
#pragma unroll
  for (int i = 0; i < 4; i++) {
    const int d = t + i * 128;
    const float y = x[i] * rn;
    buf[d] = y; c[p * 512 + d] = y;
  }
  __syncthreads();

  for (int j = 0; j < 20; j++) {     // j = k*10 + m
    float a = 0.f;
#pragma unroll
    for (int i = 0; i < 4; i++) { const int d = t + i * 128; a += buf[d] * protosn[j * 512 + d]; }
    a = reduceN(a, red, t, 128);
    if (t == 0) mk[j] = a;
  }
  __syncthreads();

  if (t == 0) {
    for (int j = 0; j < 20; j++)
      masks[p * 20 + (j % 10) * 2 + (j / 10)] = mk[j];
    float s0 = mk[0], s1 = mk[10];
    for (int m = 1; m < 10; m++) { s0 = fmaxf(s0, mk[m]); s1 = fmaxf(s1, mk[10 + m]); }
    const float mn = 0.5f * (s0 + s1);
    const float vr = 0.5f * ((s0 - mn) * (s0 - mn) + (s1 - mn) * (s1 - mn));
    const float rs2 = rsqrtf(vr + 1e-5f);
    const float o0 = (s0 - mn) * rs2 * ln2g[0] + ln2b[0];
    const float o1 = (s1 - mn) * rs2 * ln2g[1] + ln2b[1];
    const long long n = p / HW, rem = p % HW;
    outseg[(n * 2 + 0) * HW + rem] = o0;
    outseg[(n * 2 + 1) * HW + rem] = o1;
    const int pred = (o1 > o0) ? 1 : 0;
    correct[p] = (gt[p] == pred) ? 1.f : 0.f;
  }
}

// -------------------------------------------------------- masked sinkhorn
// scal layout (floats): [0..1]=Bn per class, [4..13]=rs[10],
// [16..75]=rowsums S[class][iter][10], [80..99]=nvec[class][10]
__global__ __launch_bounds__(256) void sk_base_kernel(
    const float* __restrict__ masks, const int* __restrict__ gt,
    float* __restrict__ base, float* __restrict__ cs, float* __restrict__ scal, int k) {
  __shared__ float red[256];
  const long long p = (long long)blockIdx.x * 256 + threadIdx.x;
  const int cls = (gt[p] == k);
#pragma unroll
  for (int m = 0; m < 10; m++)
    base[p * 10 + m] = cls ? __expf(masks[p * 20 + m * 2 + k] * 20.0f) : 0.0f;  // 1/eps = 20
  cs[p] = 1.0f;
  float v = cls ? 1.f : 0.f;
  v = reduceN(v, red, threadIdx.x, 256);
  if (threadIdx.x == 0) atomicAdd(&scal[k], v);
}

__global__ __launch_bounds__(256) void sk_rowsum_kernel(
    const float* __restrict__ base, const float* __restrict__ cs,
    float* __restrict__ scal, int k, int iter) {
  __shared__ float red[256];
  const long long p = (long long)blockIdx.x * 256 + threadIdx.x;
  const float cw = cs[p];
  for (int m = 0; m < 10; m++) {
    float v = base[p * 10 + m] * cw;
    v = reduceN(v, red, threadIdx.x, 256);
    if (threadIdx.x == 0) atomicAdd(&scal[16 + (k * 3 + iter) * 10 + m], v);
  }
}

__global__ void sk_rowupd_kernel(float* scal, int k, int iter) {
  const int t = threadIdx.x;
  if (t < 10) {
    const float prev = (iter == 0) ? 1.0f : scal[4 + t];
    const float rsum = prev * scal[16 + (k * 3 + iter) * 10 + t];
    scal[4 + t] = prev / (fmaxf(rsum, 1e-12f) * 10.0f);
  }
}

__global__ __launch_bounds__(256) void sk_colupd_kernel(
    const float* __restrict__ base, float* __restrict__ cs, const float* __restrict__ scal) {
  const long long p = (long long)blockIdx.x * 256 + threadIdx.x;
  float colfac = 0.f;
#pragma unroll
  for (int m = 0; m < 10; m++) colfac += base[p * 10 + m] * scal[4 + m];
  if (colfac > 0.f) {
    const float c0 = cs[p];
    cs[p] = c0 / fmaxf(c0 * colfac, 1e-12f);
  }
}

__global__ __launch_bounds__(256) void sk_mq_kernel(
    float* __restrict__ base, const float* __restrict__ cs,
    const float* __restrict__ correct, float* __restrict__ scal, int k) {
  __shared__ float red[256];
  const long long p = (long long)blockIdx.x * 256 + threadIdx.x;
  const float w = correct[p] * cs[p];
  for (int m = 0; m < 10; m++) {
    const float mq = base[p * 10 + m] * scal[4 + m] * w;
    base[p * 10 + m] = mq;
    float v = reduceN(mq, red, threadIdx.x, 256);
    if (threadIdx.x == 0) atomicAdd(&scal[80 + k * 10 + m], v);
  }
}

__global__ __launch_bounds__(256) void sk_f_kernel(
    const float* __restrict__ mq, const float* __restrict__ c,
    float* __restrict__ f, int k) {
  const int d = blockIdx.x * 256 + threadIdx.x;   // grid.x = 2
  const int m = blockIdx.y;                       // 10
  const long long p0 = (long long)blockIdx.z * HW;// grid.z = 8
  float acc = 0.f;
  for (long long p = p0; p < p0 + HW; ++p)
    acc += mq[p * 10 + m] * c[p * 512 + d];
  atomicAdd(&f[(k * 10 + m) * 512 + d], acc);
}

__global__ __launch_bounds__(128) void proto_final_kernel(
    const float* __restrict__ protosn, const float* __restrict__ f,
    const float* __restrict__ scal, float* __restrict__ outp) {
  __shared__ float red[128];
  const int j = blockIdx.x, k = j / 10, t = threadIdx.x;
  float v[4]; float ss = 0.f;
#pragma unroll
  for (int i = 0; i < 4; i++) { v[i] = f[j * 512 + t + i * 128]; ss += v[i] * v[i]; }
  ss = reduceN(ss, red, t, 128);
  const float rn = 1.f / fmaxf(sqrtf(ss), 1e-12f);
  const float nv = scal[80 + j];
  float nsum = 0.f;
  for (int m = 0; m < 10; m++) nsum += scal[80 + k * 10 + m];
  const float bn = scal[k];
  const int cond = (nv > 0.f) && (nsum > 0.f) && (bn > 0.f);
  float u[4]; float us = 0.f;
#pragma unroll
  for (int i = 0; i < 4; i++) {
    const float pv = protosn[j * 512 + t + i * 128];
    const float uu = cond ? (0.999f * pv + 0.001f * (v[i] * rn)) : pv;
    u[i] = uu; us += uu * uu;
  }
  us = reduceN(us, red, t, 128);
  const float rn2 = 1.f / fmaxf(sqrtf(us), 1e-12f);
#pragma unroll
  for (int i = 0; i < 4; i++) outp[j * 512 + t + i * 128] = u[i] * rn2;
}

// ---------------------------------------------------------------- launcher
extern "C" void kernel_launch(void* const* d_in, const int* in_sizes, int n_in,
                              void* d_out, int out_size, void* d_ws, size_t ws_size,
                              hipStream_t stream) {
  (void)in_sizes; (void)n_in; (void)out_size; (void)ws_size;
  const float* x1   = (const float*)d_in[0];
  const float* x2   = (const float*)d_in[1];
  const float* x3   = (const float*)d_in[2];
  const float* x4   = (const float*)d_in[3];
  const int*   gt   = (const int*)d_in[4];
  const float* dww  = (const float*)d_in[5];
  const float* pw_w = (const float*)d_in[6];
  const float* bn_g = (const float*)d_in[7];
  const float* bn_b = (const float*)d_in[8];
  const float* p1_w = (const float*)d_in[9];
  const float* p1_b = (const float*)d_in[10];
  const float* p2_w = (const float*)d_in[11];
  const float* p2_b = (const float*)d_in[12];
  const float* ln1g = (const float*)d_in[13];
  const float* ln1b = (const float*)d_in[14];
  const float* ln2g = (const float*)d_in[15];
  const float* ln2b = (const float*)d_in[16];
  const float* prot = (const float*)d_in[17];

  char* ws = (char*)d_ws;
  const size_t NB = (size_t)N_PIX;
  const size_t o_dw   = 0;                         // bf16 N x 960 (later reused as t2)
  const size_t o_t1   = o_dw  + NB * 960 * 2;      // bf16 N x 512
  const size_t o_t2   = o_dw;                      // alias (dw dead after GEMM1)
  const size_t o_c    = o_t1  + NB * 512 * 2;      // f32  N x 512 (p2 out, then _c in place)
  const size_t o_wpw  = o_c   + NB * 512 * 4;      // bf16 512 x 960
  const size_t o_wp1  = o_wpw + 512 * 960 * 2;     // bf16 512 x 512
  const size_t o_wp2  = o_wp1 + 512 * 512 * 2;     // bf16 512 x 512
  const size_t o_pn   = o_wp2 + 512 * 512 * 2;     // f32  20 x 512
  const size_t o_msk  = o_pn  + 20 * 512 * 4;      // f32  N x 20
  const size_t o_cor  = o_msk + NB * 20 * 4;       // f32  N
  const size_t o_base = o_cor + NB * 4;            // f32  N x 10
  const size_t o_cs   = o_base + NB * 10 * 4;      // f32  N
  const size_t o_scal = o_cs  + NB * 4;            // f32  128 scalars + 2*10*512 f

  unsigned short* dw  = (unsigned short*)(ws + o_dw);
  unsigned short* t1  = (unsigned short*)(ws + o_t1);
  unsigned short* t2  = (unsigned short*)(ws + o_t2);
  float* cbuf         = (float*)(ws + o_c);
  unsigned short* wpw = (unsigned short*)(ws + o_wpw);
  unsigned short* wp1 = (unsigned short*)(ws + o_wp1);
  unsigned short* wp2 = (unsigned short*)(ws + o_wp2);
  float* protosn      = (float*)(ws + o_pn);
  float* masks        = (float*)(ws + o_msk);
  float* correct      = (float*)(ws + o_cor);
  float* base         = (float*)(ws + o_base);
  float* cs           = (float*)(ws + o_cs);
  float* scal         = (float*)(ws + o_scal);
  float* fbuf         = scal + 128;

  float* out_seg    = (float*)d_out;               // (8,2,112,112)
  float* out_protos = out_seg + 8 * 2 * HW;        // (2,10,512)

  // 0) init: zero accumulators, convert weights, normalize prototypes
  zero_kernel<<<41, 256, 0, stream>>>(scal, 128 + 20 * 512);
  cvt_bf16_kernel<<<1920, 256, 0, stream>>>(wpw, pw_w, 512 * 960);
  cvt_bf16_kernel<<<1024, 256, 0, stream>>>(wp1, p1_w, 512 * 512);
  cvt_bf16_kernel<<<1024, 256, 0, stream>>>(wp2, p2_w, 512 * 512);
  protos_norm_kernel<<<20, 128, 0, stream>>>(prot, protosn);

  // 1) depthwise 3x3 over virtual multi-scale concat -> bf16 A matrix (N x 960)
  dw_kernel<<<dim3(3136, 120), dim3(32, 8), 0, stream>>>(x1, x2, x3, x4, dww, dw);

  // 2) WMMA GEMM chain
  gemm_bf16_kernel<0><<<dim3(784, 8), 256, 0, stream>>>(dw, wpw, 960, bn_g, bn_b, t1, nullptr);
  gemm_bf16_kernel<1><<<dim3(784, 8), 256, 0, stream>>>(t1, wp1, 512, nullptr, p1_b, t2, nullptr);
  gemm_bf16_kernel<2><<<dim3(784, 8), 256, 0, stream>>>(t2, wp2, 512, nullptr, p2_b, nullptr, cbuf);

  // 3) LN + L2 + masks + seg + pred/correct
  ln_masks_seg_kernel<<<100352, 128, 0, stream>>>(cbuf, ln1g, ln1b, protosn, masks,
                                                  out_seg, ln2g, ln2b, gt, correct);

  // 4) per-class masked sinkhorn + prototype feature accumulation
  for (int k = 0; k < 2; ++k) {
    sk_base_kernel<<<392, 256, 0, stream>>>(masks, gt, base, cs, scal, k);
    for (int it = 0; it < 3; ++it) {
      sk_rowsum_kernel<<<392, 256, 0, stream>>>(base, cs, scal, k, it);
      sk_rowupd_kernel<<<1, 16, 0, stream>>>(scal, k, it);
      sk_colupd_kernel<<<392, 256, 0, stream>>>(base, cs, scal);
    }
    sk_mq_kernel<<<392, 256, 0, stream>>>(base, cs, correct, scal, k);
    sk_f_kernel<<<dim3(2, 10, 8), 256, 0, stream>>>(base, cbuf, fbuf, k);
  }

  // 5) EMA update + renormalize -> out_protos
  proto_final_kernel<<<20, 128, 0, stream>>>(protosn, fbuf, scal, out_protos);
}
